// MHSA_89876485636389
// MI455X (gfx1250) — matrix operand order repaired and checked by reading, not compiled, not run on hardware
//
#include <hip/hip_runtime.h>

// ---------------------------------------------------------------------------
// MHSA for MI455X (gfx1250): bf16 WMMA, flash attention, b128 LDS data paths,
// async global->LDS copies (ASYNCcnt) via gfx1250 builtins when available.
// ---------------------------------------------------------------------------

constexpr int D_MODEL = 1024;
constexpr int N_SEQ   = 2048;
constexpr int HEADS   = 16;
constexpr int D_HEAD  = 64;

typedef __attribute__((ext_vector_type(16))) __bf16          v16bf;
typedef __attribute__((ext_vector_type(8)))  float           v8f;
typedef __attribute__((ext_vector_type(8)))  unsigned short  ushort8;  // 16 bytes
typedef int v4i __attribute__((vector_size(16)));                      // 16 bytes

union BFrag { v16bf v; ushort8 h[2]; unsigned short u[16]; };

// ---- CDNA5 async global->LDS path (guarded; falls back to b128 ld/st) ------
#if defined(__has_builtin)
#if __has_builtin(__builtin_amdgcn_global_load_async_to_lds_b128) && \
    __has_builtin(__builtin_amdgcn_s_wait_asynccnt)
#define ASYNC_LDS 1
#endif
#endif
#ifndef ASYNC_LDS
#define ASYNC_LDS 0
#endif

__device__ __forceinline__ void copy16_g2l(unsigned short* lds_dst,
                                           const unsigned short* gsrc) {
#if ASYNC_LDS
  // param0: addrspace(1) v4i*  (printed as "int __vector(4) __device__ *")
  // param1: addrspace(3) v4i*  (LDS destination)
  __builtin_amdgcn_global_load_async_to_lds_b128(
      (__attribute__((address_space(1))) v4i*)gsrc,
      (__attribute__((address_space(3))) v4i*)lds_dst, 0, 0);
#else
  *(ushort8*)lds_dst = *(const ushort8*)gsrc;
#endif
}

__device__ __forceinline__ void async_join() {
#if ASYNC_LDS
  __builtin_amdgcn_s_wait_asynccnt(0);  // all async LDS writes by this wave done
#endif
}

__device__ __forceinline__ unsigned short f2bf(float f) {
  union { float f; unsigned u; } c; c.f = f;
  unsigned r = 0x7FFFu + ((c.u >> 16) & 1u);   // round-to-nearest-even
  return (unsigned short)((c.u + r) >> 16);
}

__device__ __forceinline__ v8f v8f_zero() {
  v8f v;
#pragma unroll
  for (int i = 0; i < 8; ++i) v[i] = 0.0f;
  return v;
}

// A fragment (16x32 MxK bf16). Tile stored row-major [m][k], ld % 8 == 0,
// 16B-aligned. Lane layout (ISA): u[0..7] = A[m][hi*8 .. hi*8+7],
// u[8..15] = A[m][16+hi*8 .. 16+hi*8+7]  ->  two ds_load_b128.
__device__ __forceinline__ v16bf load_a_frag(const unsigned short* src, int ld, int lane) {
  const int m  = lane & 15;
  const int hi = (lane >> 4) & 1;
  BFrag f;
  f.h[0] = *(const ushort8*)(src + m * ld + hi * 8);
  f.h[1] = *(const ushort8*)(src + m * ld + 16 + hi * 8);
  return f.v;
}

// B fragment (32x16 KxN bf16). Tile stored TRANSPOSED [n][k] (k contiguous),
// ld % 8 == 0. Lane layout: u[0..7] = B[kb .. kb+7][n], u[8..15] = B[16+kb .. 16+kb+7][n]  ->  two b128.
__device__ __forceinline__ v16bf load_b_frag(const unsigned short* src, int ld, int lane) {
  const int n  = lane & 15;
  const int kb = ((lane >> 4) & 1) * 8;
  BFrag f;
  f.h[0] = *(const ushort8*)(src + n * ld + kb);
  f.h[1] = *(const ushort8*)(src + n * ld + kb + 16);
  return f.v;
}

#define WMMA_BF16(a, b, c) \
  __builtin_amdgcn_wmma_f32_16x16x32_bf16(false, (a), false, (b), (short)0, (c), false, false)

// ---------------------------------------------------------------------------
// 1) global mean / mean-square reduction
// ---------------------------------------------------------------------------
__global__ void k_stats_zero(float* stats) { stats[0] = 0.0f; stats[1] = 0.0f; }

__global__ __launch_bounds__(256) void k_reduce(const float* __restrict__ x,
                                                float* __restrict__ stats, int n) {
  __shared__ float s_sum[256];
  __shared__ float s_sq[256];
  float a = 0.0f, b = 0.0f;
  for (int i = blockIdx.x * blockDim.x + threadIdx.x; i < n; i += gridDim.x * blockDim.x) {
    float v = x[i];
    a += v;
    b += v * v;
  }
  s_sum[threadIdx.x] = a;
  s_sq[threadIdx.x]  = b;
  __syncthreads();
  for (int s = 128; s > 0; s >>= 1) {
    if ((int)threadIdx.x < s) {
      s_sum[threadIdx.x] += s_sum[threadIdx.x + s];
      s_sq[threadIdx.x]  += s_sq[threadIdx.x + s];
    }
    __syncthreads();
  }
  if (threadIdx.x == 0) {
    atomicAdd(&stats[0], s_sum[0]);
    atomicAdd(&stats[1], s_sq[0]);
  }
}

// ---------------------------------------------------------------------------
// 2) normalize -> bf16 transposed xnT[n][c] ; weight conversions
// ---------------------------------------------------------------------------
__global__ void k_normalize_t(const float* __restrict__ x, const float* __restrict__ stats,
                              unsigned short* __restrict__ xnT, int n) {
  const float inv_n = 1.0f / (float)n;
  const float mean  = stats[0] * inv_n;
  const float var   = stats[1] * inv_n - mean * mean;
  const float rs    = rsqrtf(var + 1e-5f);
  int i = blockIdx.x * blockDim.x + threadIdx.x;  // i = c*N_SEQ + nn
  if (i < n) {
    int c  = i >> 11;          // / N_SEQ
    int nn = i & (N_SEQ - 1);  // % N_SEQ
    xnT[(size_t)nn * D_MODEL + c] = f2bf((x[i] - mean) * rs);
  }
}

// WQ/WK/WV: src [H][D_MODEL][D_HEAD] f32 -> dst [H][D_HEAD][D_MODEL] bf16 (W^T)
__global__ void k_convert_wT(const float* __restrict__ src, unsigned short* __restrict__ dst,
                             int n) {
  int i = blockIdx.x * blockDim.x + threadIdx.x;
  if (i < n) {
    int h   = i >> 16;          // / (1024*64)
    int rem = i & 65535;
    int c   = rem >> 6;         // / 64
    int dh  = rem & 63;
    dst[(size_t)h * D_HEAD * D_MODEL + (size_t)dh * D_MODEL + c] = f2bf(src[i]);
  }
}

__global__ void k_convert(const float* __restrict__ src, unsigned short* __restrict__ dst, int n) {
  int i = blockIdx.x * blockDim.x + threadIdx.x;
  if (i < n) dst[i] = f2bf(src[i]);
}

// ---------------------------------------------------------------------------
// 3) QKV projection: out[h] = W[h]^T (64x1024) * xn (1024xN).
//    grid.x = HEADS*(N/128), grid.y in {0,1,2} = Q/K/V. 256 threads (8 waves).
//    Q,K written transposed [N][dH] (for flash A/B fragments); V as [dH][N].
// ---------------------------------------------------------------------------
__global__ __launch_bounds__(256) void k_qkv_gemm(
    const unsigned short* __restrict__ xnT,
    const unsigned short* __restrict__ wqT, const unsigned short* __restrict__ wkT,
    const unsigned short* __restrict__ wvT,
    unsigned short* __restrict__ qT, unsigned short* __restrict__ kT,
    unsigned short* __restrict__ vb) {
  constexpr int LDA = 40, LDB = 40;
  __shared__ __align__(16) unsigned short As[64 * LDA];    // A[m=dH][k]
  __shared__ __align__(16) unsigned short Bs[128 * LDB];   // B^T[n][k]
  const int nblocks = N_SEQ / 128;
  const int h  = blockIdx.x / nblocks;
  const int n0 = (blockIdx.x % nblocks) * 128;
  const unsigned short* WT =
      (blockIdx.y == 0 ? wqT : (blockIdx.y == 1 ? wkT : wvT)) + (size_t)h * D_HEAD * D_MODEL;
  unsigned short* Out =
      (blockIdx.y == 0 ? qT : (blockIdx.y == 1 ? kT : vb)) + (size_t)h * D_HEAD * N_SEQ;

  const int tid = threadIdx.x, lane = tid & 31, w = tid >> 5;
  const int mt = (w & 3) * 16;  // m-tile origin within dH
  const int nh = (w >> 2) * 4;  // first n-tile index (0 or 4)

  v8f acc[4];
#pragma unroll
  for (int t = 0; t < 4; ++t) acc[t] = v8f_zero();

  for (int k0 = 0; k0 < D_MODEL; k0 += 32) {
    {  // As[m][k] = WT[m][k0+k] : 64 rows x 4 b128 chunks, 1 per thread
      int m = tid >> 2, c = tid & 3;
      copy16_g2l(&As[m * LDA + c * 8], &WT[(size_t)m * D_MODEL + k0 + c * 8]);
    }
#pragma unroll
    for (int i = tid; i < 128 * 4; i += 256) {  // Bs[n][k] = xnT[n0+n][k0+k]
      int n = i >> 2, c = i & 3;
      copy16_g2l(&Bs[n * LDB + c * 8], &xnT[(size_t)(n0 + n) * D_MODEL + k0 + c * 8]);
    }
    if (k0 + 32 < D_MODEL)  // -> global_prefetch_b8
      __builtin_prefetch(&xnT[(size_t)(n0 + (tid & 127)) * D_MODEL + k0 + 32], 0, 1);
    async_join();
    __syncthreads();

    v16bf a = load_a_frag(&As[mt * LDA], LDA, lane);
#pragma unroll
    for (int t = 0; t < 4; ++t) {
      v16bf b = load_b_frag(&Bs[(nh + t) * 16 * LDB], LDB, lane);
      acc[t]  = WMMA_BF16(a, b, acc[t]);
    }
    __syncthreads();
  }

  const int hi = lane >> 4, ln = lane & 15;
  if (blockIdx.y == 2) {  // V: [dH][N], j contiguous
#pragma unroll
    for (int t = 0; t < 4; ++t)
#pragma unroll
      for (int r = 0; r < 8; ++r) {
        int row = mt + r + 8 * hi;
        int col = n0 + (nh + t) * 16 + ln;
        Out[(size_t)row * N_SEQ + col] = f2bf(acc[t][r]);
      }
  } else {  // Q,K transposed: [N][dH], dH contiguous
#pragma unroll
    for (int t = 0; t < 4; ++t)
#pragma unroll
      for (int r = 0; r < 8; ++r) {
        int row = mt + r + 8 * hi;
        int col = n0 + (nh + t) * 16 + ln;
        Out[(size_t)col * D_HEAD + row] = f2bf(acc[t][r]);
      }
  }
}

// ---------------------------------------------------------------------------
// 4) Flash attention. Block = (head, 64 queries), 4 waves; each wave owns 16
//    queries, streams 32 keys/values per step, online softmax in registers.
//    QT/KT are [N][dH] (contraction dH contiguous); V is [dH][N].
// ---------------------------------------------------------------------------
__global__ __launch_bounds__(128) void k_flash(
    const unsigned short* __restrict__ qT, const unsigned short* __restrict__ kT,
    const unsigned short* __restrict__ vb, unsigned short* __restrict__ attnT) {
  constexpr int LDQ = 72, LDK = 72, LDV = 40, LDP = 40;
  __shared__ __align__(16) unsigned short Qs[64 * LDQ];     // A tile [q][d]
  __shared__ __align__(16) unsigned short Ks[32 * LDK];     // B^T tile [j][d]
  __shared__ __align__(16) unsigned short Vs[64 * LDV];     // B^T tile [d][j]
  __shared__ __align__(16) unsigned short Ps[4][16 * LDP];  // per-wave A tile [q][j]

  const int qblocks = N_SEQ / 64;
  const int h  = blockIdx.x / qblocks;
  const int q0 = (blockIdx.x % qblocks) * 64;
  const unsigned short* Q = qT + (size_t)h * D_HEAD * N_SEQ;  // [N][64]
  const unsigned short* K = kT + (size_t)h * D_HEAD * N_SEQ;  // [N][64]
  const unsigned short* V = vb + (size_t)h * D_HEAD * N_SEQ;  // [64][N]

  const int tid = threadIdx.x, lane = tid & 31, w = tid >> 5;
  const int hi = lane >> 4, ln = lane & 15;

#pragma unroll
  for (int i = tid; i < 64 * 8; i += 128) {  // Qs[q][d] = Q[q0+q][d], b128 chunks
    int q = i >> 3, c = i & 7;
    copy16_g2l(&Qs[q * LDQ + c * 8], &Q[(size_t)(q0 + q) * D_HEAD + c * 8]);
  }
  async_join();
  __syncthreads();

  const v16bf aq0 = load_a_frag(&Qs[(w * 16) * LDQ + 0],  LDQ, lane);  // d 0..31
  const v16bf aq1 = load_a_frag(&Qs[(w * 16) * LDQ + 32], LDQ, lane);  // d 32..63

  v8f O0 = v8f_zero(), O1 = v8f_zero(), O2 = v8f_zero(), O3 = v8f_zero();
  float mrun[8], lrun[8];
#pragma unroll
  for (int r = 0; r < 8; ++r) { mrun[r] = -3.0e30f; lrun[r] = 0.0f; }

  for (int j0 = 0; j0 < N_SEQ; j0 += 32) {
#pragma unroll
    for (int i = tid; i < 32 * 8; i += 128) {  // Ks[j][d] = K[j0+j][d]
      int j = i >> 3, c = i & 7;
      copy16_g2l(&Ks[j * LDK + c * 8], &K[(size_t)(j0 + j) * D_HEAD + c * 8]);
    }
#pragma unroll
    for (int i = tid; i < 64 * 4; i += 128) {  // Vs[d][j] = V[d][j0+j]
      int d = i >> 2, c = i & 3;
      copy16_g2l(&Vs[d * LDV + c * 8], &V[(size_t)d * N_SEQ + j0 + c * 8]);
    }
    if (j0 + 32 < N_SEQ)
      __builtin_prefetch(&K[(size_t)(j0 + 32 + (tid & 31)) * D_HEAD], 0, 1);
    async_join();
    __syncthreads();

    // S = Q K^T over dH=64 (two 32-wide K-chunks), two 16-wide j tiles
    v16bf b00 = load_b_frag(&Ks[0 * LDK + 0],   LDK, lane);  // j 0..15,  d 0..31
    v16bf b01 = load_b_frag(&Ks[0 * LDK + 32],  LDK, lane);  // j 0..15,  d 32..63
    v16bf b10 = load_b_frag(&Ks[16 * LDK + 0],  LDK, lane);  // j 16..31, d 0..31
    v16bf b11 = load_b_frag(&Ks[16 * LDK + 32], LDK, lane);  // j 16..31, d 32..63
    v8f s0 = WMMA_BF16(aq0, b00, v8f_zero());
    s0     = WMMA_BF16(aq1, b01, s0);
    v8f s1 = WMMA_BF16(aq0, b10, v8f_zero());
    s1     = WMMA_BF16(aq1, b11, s1);

#pragma unroll
    for (int r = 0; r < 8; ++r) { s0[r] *= 0.125f; s1[r] *= 0.125f; }  // 1/sqrt(dH)

    // Online softmax; rows live in 16-lane column groups (C/D layout: M=r+8*hi)
#pragma unroll
    for (int r = 0; r < 8; ++r) {
      float vmax = fmaxf(s0[r], s1[r]);
#pragma unroll
      for (int off = 1; off <= 8; off <<= 1) vmax = fmaxf(vmax, __shfl_xor(vmax, off, 32));
      float mnew = fmaxf(mrun[r], vmax);
      float sc   = __expf(mrun[r] - mnew);
      float p0   = __expf(s0[r] - mnew);
      float p1   = __expf(s1[r] - mnew);
      float rs   = p0 + p1;
#pragma unroll
      for (int off = 1; off <= 8; off <<= 1) rs += __shfl_xor(rs, off, 32);
      lrun[r] = lrun[r] * sc + rs;
      mrun[r] = mnew;
      O0[r] *= sc; O1[r] *= sc; O2[r] *= sc; O3[r] *= sc;
      Ps[w][(r + 8 * hi) * LDP + ln]      = f2bf(p0);
      Ps[w][(r + 8 * hi) * LDP + 16 + ln] = f2bf(p1);
    }
    __syncthreads();  // Ps visible before A-fragment reload

    // O += P V^T (K-dim = 32 keys)
    v16bf ap = load_a_frag(&Ps[w][0], LDP, lane);
    v16bf bv;
    bv = load_b_frag(&Vs[0 * 16 * LDV], LDV, lane); O0 = WMMA_BF16(ap, bv, O0);
    bv = load_b_frag(&Vs[1 * 16 * LDV], LDV, lane); O1 = WMMA_BF16(ap, bv, O1);
    bv = load_b_frag(&Vs[2 * 16 * LDV], LDV, lane); O2 = WMMA_BF16(ap, bv, O2);
    bv = load_b_frag(&Vs[3 * 16 * LDV], LDV, lane); O3 = WMMA_BF16(ap, bv, O3);
    __syncthreads();  // before Ks/Vs are overwritten
  }

  // attnT[q][h*64 + d] = O[q][d] / l[q]   (d contiguous across lanes)
#pragma unroll
  for (int r = 0; r < 8; ++r) {
    float inv = 1.0f / lrun[r];
    int q = q0 + w * 16 + r + 8 * hi;
    unsigned short* dst = attnT + (size_t)q * D_MODEL + h * D_HEAD;
    dst[0 * 16 + ln] = f2bf(O0[r] * inv);
    dst[1 * 16 + ln] = f2bf(O1[r] * inv);
    dst[2 * 16 + ln] = f2bf(O2[r] * inv);
    dst[3 * 16 + ln] = f2bf(O3[r] * inv);
  }
}

// ---------------------------------------------------------------------------
// 5) out = W0 @ attn + x   (f32 output, residual add). attnT is [N][D].
// ---------------------------------------------------------------------------
__global__ __launch_bounds__(256) void k_outproj(
    const unsigned short* __restrict__ w0, const unsigned short* __restrict__ attnT,
    const float* __restrict__ x, float* __restrict__ out) {
  constexpr int LDA = 40, LDB = 40;
  __shared__ __align__(16) unsigned short As[64 * LDA];   // A[m][k] = W0 tile
  __shared__ __align__(16) unsigned short Bs[128 * LDB];  // B^T[n][k] = attn^T tile
  const int nblocks = N_SEQ / 128;
  const int m0 = (blockIdx.x / nblocks) * 64;
  const int n0 = (blockIdx.x % nblocks) * 128;

  const int tid = threadIdx.x, lane = tid & 31, w = tid >> 5;
  const int mt = (w & 3) * 16;
  const int nh = (w >> 2) * 4;

  v8f acc[4];
#pragma unroll
  for (int t = 0; t < 4; ++t) acc[t] = v8f_zero();

  for (int k0 = 0; k0 < D_MODEL; k0 += 32) {
    {  // As[m][k] = W0[m0+m][k0+k]
      int m = tid >> 2, c = tid & 3;
      copy16_g2l(&As[m * LDA + c * 8], &w0[(size_t)(m0 + m) * D_MODEL + k0 + c * 8]);
    }
#pragma unroll
    for (int i = tid; i < 128 * 4; i += 256) {  // Bs[n][k] = attnT[n0+n][k0+k]
      int n = i >> 2, c = i & 3;
      copy16_g2l(&Bs[n * LDB + c * 8], &attnT[(size_t)(n0 + n) * D_MODEL + k0 + c * 8]);
    }
    if (k0 + 32 < D_MODEL)
      __builtin_prefetch(&attnT[(size_t)(n0 + (tid & 127)) * D_MODEL + k0 + 32], 0, 1);
    async_join();
    __syncthreads();

    v16bf a = load_a_frag(&As[mt * LDA], LDA, lane);
#pragma unroll
    for (int t = 0; t < 4; ++t) {
      v16bf b = load_b_frag(&Bs[(nh + t) * 16 * LDB], LDB, lane);
      acc[t]  = WMMA_BF16(a, b, acc[t]);
    }
    __syncthreads();
  }

  const int hi = lane >> 4, ln = lane & 15;
#pragma unroll
  for (int t = 0; t < 4; ++t)
#pragma unroll
    for (int r = 0; r < 8; ++r) {
      int row = m0 + mt + r + 8 * hi;
      int col = n0 + (nh + t) * 16 + ln;
      out[(size_t)row * N_SEQ + col] = acc[t][r] + x[(size_t)row * N_SEQ + col];
    }
}

// ---------------------------------------------------------------------------
// launch
// ---------------------------------------------------------------------------
extern "C" void kernel_launch(void* const* d_in, const int* in_sizes, int n_in,
                              void* d_out, int out_size, void* d_ws, size_t ws_size,
                              hipStream_t stream) {
  const float* x  = (const float*)d_in[0];
  const float* WQ = (const float*)d_in[1];
  const float* WK = (const float*)d_in[2];
  const float* WV = (const float*)d_in[3];
  const float* W0 = (const float*)d_in[4];
  float* out = (float*)d_out;

  char* ws = (char*)d_ws;
  size_t off = 0;
  auto take = [&](size_t bytes) -> char* {
    char* p = ws + off;
    off += (bytes + 255) & ~(size_t)255;
    return p;
  };

  const size_t xn_elems = (size_t)D_MODEL * N_SEQ;            // 2M
  const size_t w_elems  = (size_t)HEADS * D_MODEL * D_HEAD;   // 1M per tensor
  const size_t w0_elems = (size_t)D_MODEL * D_MODEL;          // 1M
  const size_t p_elems  = (size_t)HEADS * D_HEAD * N_SEQ;     // 2M per tensor

  float*          stats = (float*)take(2 * sizeof(float));
  unsigned short* xnT   = (unsigned short*)take(xn_elems * 2);  // [N][D]
  unsigned short* wqT   = (unsigned short*)take(w_elems * 2);   // [H][dH][D]
  unsigned short* wkT   = (unsigned short*)take(w_elems * 2);
  unsigned short* wvT   = (unsigned short*)take(w_elems * 2);
  unsigned short* w0b   = (unsigned short*)take(w0_elems * 2);  // [D][D]
  unsigned short* qT    = (unsigned short*)take(p_elems * 2);   // [H][N][dH]
  unsigned short* kT    = (unsigned short*)take(p_elems * 2);   // [H][N][dH]
  unsigned short* vb    = (unsigned short*)take(p_elems * 2);   // [H][dH][N]
  unsigned short* attnT = (unsigned short*)take(xn_elems * 2);  // [N][D]
  (void)ws_size; (void)in_sizes; (void)n_in; (void)out_size;

  const int n = (int)xn_elems;

  k_stats_zero<<<1, 1, 0, stream>>>(stats);
  k_reduce<<<1024, 256, 0, stream>>>(x, stats, n);
  k_normalize_t<<<(n + 255) / 256, 256, 0, stream>>>(x, stats, xnT, n);
  k_convert_wT<<<(int)((w_elems + 255) / 256), 256, 0, stream>>>(WQ, wqT, (int)w_elems);
  k_convert_wT<<<(int)((w_elems + 255) / 256), 256, 0, stream>>>(WK, wkT, (int)w_elems);
  k_convert_wT<<<(int)((w_elems + 255) / 256), 256, 0, stream>>>(WV, wvT, (int)w_elems);
  k_convert<<<(int)((w0_elems + 255) / 256), 256, 0, stream>>>(W0, w0b, (int)w0_elems);

  dim3 gqkv(HEADS * (N_SEQ / 128), 3);
  k_qkv_gemm<<<gqkv, 256, 0, stream>>>(xnT, wqT, wkT, wvT, qT, kT, vb);

  k_flash<<<HEADS * (N_SEQ / 64), 128, 0, stream>>>(qT, kT, vb, attnT);

  k_outproj<<<(D_MODEL / 64) * (N_SEQ / 128), 256, 0, stream>>>(w0b, attnT, x, out);
}
